// NeuralRadianceField_87668872446725
// MI455X (gfx1250) — matrix operand order repaired and checked by reading, not compiled, not run on hardware
//
#include <hip/hip_runtime.h>
#include <hip/hip_bf16.h>

typedef __attribute__((ext_vector_type(16))) _Float16 v16h;
typedef __attribute__((ext_vector_type(8)))  float    v8f;
typedef __attribute__((ext_vector_type(4)))  unsigned int v4u;
typedef __attribute__((ext_vector_type(4)))  unsigned int u32x4;
typedef __attribute__((ext_vector_type(8)))  int i32x8;
typedef __attribute__((ext_vector_type(4)))  int i32x4;

#if defined(__has_builtin)
#if __has_builtin(__builtin_amdgcn_tensor_load_to_lds) && __has_builtin(__builtin_amdgcn_s_wait_tensorcnt)
#define HAVE_TDM 1
#endif
#endif
#ifndef HAVE_TDM
#define HAVE_TDM 0
#endif

// ---- weight image layout (offsets in _Float16 elements); identical in d_ws and LDS ----
#define W0T_OFF    0        // [128][64]   w0^T, K padded 39->64
#define W1T_OFF    8192     // [128][128]
#define W2T_OFF    24576    // [128][128]
#define W3AT_OFF   40960    // [128][128]  w3 rows 0..127 (y part)
#define W3BT_OFF   57344    // [128][64]   w3 rows 128..166 (xyz part), K pad 39->64
#define W4T_OFF    65536    // [128][128]
#define W5T_OFF    81920    // [128][128]
#define WDT_OFF    98304    // [144][128]  wd^T, N padded 129->144
#define WR1AT_OFF  116736   // [64][128]   wr1 rows 0..127 (x[:,1:])
#define WR1BT_OFF  124928   // [64][32]    wr1 rows 128..142 (dir), K pad 15->32
#define WR2T_OFF   126976   // [16][64]    wr2^T, N padded 3->16
#define WEIGHT_HALVES 128000

// bias offsets (floats, within bias region)
#define B0_OFF 0
#define B1_OFF 128
#define B2_OFF 256
#define B3_OFF 384
#define B4_OFF 512
#define B5_OFF 640
#define BD_OFF 768   // padded 129->144
#define BR1_OFF 912
#define BR2_OFF 976  // padded 3->16
#define SBIAS_FLOATS 992

// contiguous image: f16 weights then f32 biases (DMA'd as one tile)
#define WS_WEIGHT_BYTES (WEIGHT_HALVES * 2)            // 256,000
#define WS_BIAS_BYTE_OFF WS_WEIGHT_BYTES
#define WS_BYTES (WS_WEIGHT_BYTES + SBIAS_FLOATS * 4)  // 259,968 (= 32,496 x 8B)
#define WS_QWORDS (WS_BYTES / 8)                       // 32,496

// LDS: [image 259,968 B][ACT][XYZ][DIR]
#define ACT_OFF    129984   // halves; 4 waves x [32][128]
#define XYZ_OFF    146368   // 4 waves x [32][64]
#define DIR_OFF    154560   // 4 waves x [32][32]
#define SMEM_HALVES 158656  // 317,312 bytes total

#define THREADS 128

#define LDS_FENCE() asm volatile("s_wait_dscnt 0" ::: "memory")

// A fragment (16x32 f16): lane holds row M=lane&15.
// element e -> K = k_off + e + 8*(e>>3) + 8*(lane>>4): two contiguous 16B chunks.
__device__ __forceinline__ v16h load_a_frag(const _Float16* base, int ld, int k_off, int lane) {
  const _Float16* p = base + (lane & 15) * ld + k_off + 8 * (lane >> 4);
  union { v4u u[2]; v16h h; } t;
  t.u[0] = *(const v4u*)(p);
  t.u[1] = *(const v4u*)(p + 16);
  return t.h;
}

// B fragment (32x16 f16) from transposed weights wT[N][K] (K contiguous):
// element e -> K = k_off + e + 8*(e>>3) + 8*(lane>>4), N = n_off + (lane&15).
__device__ __forceinline__ v16h load_b_frag(const _Float16* wT, int ldk, int n_off, int k_off, int lane) {
  const _Float16* p = wT + (n_off + (lane & 15)) * ldk + k_off + 8 * (lane >> 4);
  union { v4u u[2]; v16h h; } t;
  t.u[0] = *(const v4u*)(p);
  t.u[1] = *(const v4u*)(p + 16);
  return t.h;
}

template<int NT>
__device__ __forceinline__ void init_bias(const float* bias, int lane, v8f* acc) {
#pragma unroll
  for (int n = 0; n < NT; ++n) {
    float bv = bias[n * 16 + (lane & 15)];
    v8f c;
#pragma unroll
    for (int r = 0; r < 8; ++r) c[r] = bv;
    acc[n] = c;
  }
}

// Two M-tiles (32 rows) share every B fragment -> halves LDS B traffic per WMMA.
template<int KT, int NT>
__device__ __forceinline__ void dense2(const _Float16* A0, const _Float16* A1, int lda,
                                       const _Float16* wT, int ldk,
                                       int lane, v8f* acc0, v8f* acc1) {
  v16h a0[KT], a1[KT];
#pragma unroll
  for (int k = 0; k < KT; ++k) {
    a0[k] = load_a_frag(A0, lda, k * 32, lane);
    a1[k] = load_a_frag(A1, lda, k * 32, lane);
  }
#pragma unroll
  for (int n = 0; n < NT; ++n) {
#pragma unroll
    for (int k = 0; k < KT; ++k) {
      v16h b = load_b_frag(wT, ldk, n * 16, k * 32, lane);
      acc0[n] = __builtin_amdgcn_wmma_f32_16x16x32_f16(false, a0[k], false, b,
                                                       (short)0, acc0[n], false, false);
      acc1[n] = __builtin_amdgcn_wmma_f32_16x16x32_f16(false, a1[k], false, b,
                                                       (short)0, acc1[n], false, false);
    }
  }
}

template<int NT>
__device__ __forceinline__ void store_relu(_Float16* stage, int ld, int lane, const v8f* acc) {
  int mb = 8 * (lane >> 4), nc = lane & 15;
#pragma unroll
  for (int n = 0; n < NT; ++n)
#pragma unroll
    for (int r = 0; r < 8; ++r) {
      float x = acc[n][r];
      x = x > 0.f ? x : 0.f;
      stage[(r + mb) * ld + n * 16 + nc] = (_Float16)x;
    }
}

// Generic f32 -> f16 transposed stage: dst[o*ldk+i] = src[(row_off+i)*fan_out+o]
__device__ __forceinline__ void stage_wT(_Float16* dst, int ldk, int npad,
                                         const float* __restrict__ src,
                                         int fan_out, int row_off, int rows,
                                         int tid, int stride) {
  int total = npad * ldk;
  for (int idx = tid; idx < total; idx += stride) {
    int i = idx / npad;          // input (K) index
    int o = idx - i * npad;      // output (N) index -> coalesced global read
    float v = 0.f;
    if (i < rows && o < fan_out) v = src[(row_off + i) * fan_out + o];
    dst[o * ldk + i] = (_Float16)v;
  }
}

__device__ __forceinline__ void stage_bias(float* dst, const float* __restrict__ src,
                                           int n, int npad, int tid, int stride) {
  for (int i = tid; i < npad; i += stride) dst[i] = (i < n) ? src[i] : 0.f;
}

__device__ __forceinline__ void stage_all(_Float16* wh, float* bf,
                                          const float* w0, const float* b0,
                                          const float* w1, const float* b1,
                                          const float* w2, const float* b2,
                                          const float* w3, const float* b3,
                                          const float* w4, const float* b4,
                                          const float* w5, const float* b5,
                                          const float* wd, const float* bd,
                                          const float* wr1, const float* br1,
                                          const float* wr2, const float* br2,
                                          int tid, int stride) {
  stage_wT(wh + W0T_OFF,   64, 128, w0, 128,   0,  39, tid, stride);
  stage_wT(wh + W1T_OFF,  128, 128, w1, 128,   0, 128, tid, stride);
  stage_wT(wh + W2T_OFF,  128, 128, w2, 128,   0, 128, tid, stride);
  stage_wT(wh + W3AT_OFF, 128, 128, w3, 128,   0, 128, tid, stride);
  stage_wT(wh + W3BT_OFF,  64, 128, w3, 128, 128,  39, tid, stride);
  stage_wT(wh + W4T_OFF,  128, 128, w4, 128,   0, 128, tid, stride);
  stage_wT(wh + W5T_OFF,  128, 128, w5, 128,   0, 128, tid, stride);
  stage_wT(wh + WDT_OFF,  128, 144, wd, 129,   0, 128, tid, stride);
  stage_wT(wh + WR1AT_OFF,128,  64, wr1, 64,   0, 128, tid, stride);
  stage_wT(wh + WR1BT_OFF, 32,  64, wr1, 64, 128,  15, tid, stride);
  stage_wT(wh + WR2T_OFF,  64,  16, wr2,  3,   0,  64, tid, stride);
  stage_bias(bf + B0_OFF, b0, 128, 128, tid, stride);
  stage_bias(bf + B1_OFF, b1, 128, 128, tid, stride);
  stage_bias(bf + B2_OFF, b2, 128, 128, tid, stride);
  stage_bias(bf + B3_OFF, b3, 128, 128, tid, stride);
  stage_bias(bf + B4_OFF, b4, 128, 128, tid, stride);
  stage_bias(bf + B5_OFF, b5, 128, 128, tid, stride);
  stage_bias(bf + BD_OFF, bd, 129, 144, tid, stride);
  stage_bias(bf + BR1_OFF, br1, 64, 64, tid, stride);
  stage_bias(bf + BR2_OFF, br2,  3, 16, tid, stride);
}

// ---- prep kernel: build the f16-transposed weight image in global workspace ----
__global__ __launch_bounds__(256)
void nerf_prep_kernel(const float* __restrict__ w0, const float* __restrict__ b0,
                      const float* __restrict__ w1, const float* __restrict__ b1,
                      const float* __restrict__ w2, const float* __restrict__ b2,
                      const float* __restrict__ w3, const float* __restrict__ b3,
                      const float* __restrict__ w4, const float* __restrict__ b4,
                      const float* __restrict__ w5, const float* __restrict__ b5,
                      const float* __restrict__ wd, const float* __restrict__ bd,
                      const float* __restrict__ wr1, const float* __restrict__ br1,
                      const float* __restrict__ wr2, const float* __restrict__ br2,
                      void* __restrict__ ws) {
  _Float16* wh = (_Float16*)ws;
  float* bf = (float*)((char*)ws + WS_BIAS_BYTE_OFF);
  int tid = blockIdx.x * blockDim.x + threadIdx.x;
  int stride = gridDim.x * blockDim.x;
  stage_all(wh, bf, w0, b0, w1, b1, w2, b2, w3, b3, w4, b4, w5, b5,
            wd, bd, wr1, br1, wr2, br2, tid, stride);
}

// ---- fused persistent NeRF kernel: 128 threads = 4 waves x 32 points = 1 ray/iter ----
__global__ __launch_bounds__(THREADS, 1)
void nerf_fused_kernel(const float* __restrict__ pts, const float* __restrict__ dirs,
                       const float* __restrict__ w0, const float* __restrict__ b0,
                       const float* __restrict__ w1, const float* __restrict__ b1,
                       const float* __restrict__ w2, const float* __restrict__ b2,
                       const float* __restrict__ w3, const float* __restrict__ b3,
                       const float* __restrict__ w4, const float* __restrict__ b4,
                       const float* __restrict__ w5, const float* __restrict__ b5,
                       const float* __restrict__ wd, const float* __restrict__ bd,
                       const float* __restrict__ wr1, const float* __restrict__ br1,
                       const float* __restrict__ wr2, const float* __restrict__ br2,
                       const void* __restrict__ wsrc, int from_ws,
                       float* __restrict__ out, int ntotal, int n_rays) {
  __shared__ __attribute__((aligned(16))) unsigned char sraw[SMEM_HALVES * 2];
  _Float16* smem = (_Float16*)sraw;
  float* sbias = (float*)(sraw + WS_BIAS_BYTE_OFF);

  const int tid  = threadIdx.x;
  const int wave = tid >> 5;
  const int lane = tid & 31;

  if (from_ws) {
#if HAVE_TDM
    // Single Tensor-Data-Mover DMA: 259,968B global image -> LDS (one 2D tile of 8B units)
    if (wave == 0) {
      unsigned lds_base = (unsigned)(unsigned long long)(uintptr_t)sraw;
      unsigned long long ga = (unsigned long long)(uintptr_t)wsrc;
      u32x4 g0;
      g0[0] = 1u;                                           // count=1, user descriptor
      g0[1] = lds_base;                                     // lds_addr (bytes)
      g0[2] = (unsigned)(ga & 0xFFFFFFFFu);                 // global_addr[31:0]
      g0[3] = (unsigned)((ga >> 32) & 0x01FFFFFFu) | (2u << 30); // global_addr[56:32] | type=2
      i32x8 g1;
      g1[0] = 0x00030000;                 // wg_mask=0, data_size=3 (8B)
      g1[1] = (int)((unsigned)WS_QWORDS << 16);   // tensor_dim0[15:0] in [63:48]
      g1[2] = (int)(1u << 16);            // tensor_dim0[31:16]=0, tensor_dim1[15:0]=1
      g1[3] = (int)((unsigned)WS_QWORDS << 16);   // tensor_dim1[31:16]=0, tile_dim0
      g1[4] = 1;                          // tile_dim1=1, tile_dim2=0
      g1[5] = (int)WS_QWORDS;             // tensor_dim0_stride[31:0]
      g1[6] = 0;                          // stride0[47:32]=0, stride1[15:0]=0
      g1[7] = 0;
      i32x4 gz4 = {0, 0, 0, 0};
      i32x8 gz8 = {0, 0, 0, 0, 0, 0, 0, 0};
      __builtin_amdgcn_tensor_load_to_lds(g0, g1, gz4, gz4, gz8, 0);
      __builtin_amdgcn_s_wait_tensorcnt(0);
    }
#else
    const v4u* s = (const v4u*)wsrc;
    v4u* d = (v4u*)sraw;
    for (int i = tid; i < WS_BYTES / 16; i += THREADS) d[i] = s[i];
#endif
  } else {
    stage_all(smem, sbias, w0, b0, w1, b1, w2, b2, w3, b3, w4, b4, w5, b5,
              wd, bd, wr1, br1, wr2, br2, tid, THREADS);
  }
  __syncthreads();

  _Float16* ACT0 = smem + ACT_OFF + wave * 32 * 128;
  _Float16* ACT1 = ACT0 + 16 * 128;
  _Float16* XYZ0 = smem + XYZ_OFF + wave * 32 * 64;
  _Float16* XYZ1 = XYZ0 + 16 * 64;
  _Float16* DIR0 = smem + DIR_OFF + wave * 32 * 32;
  _Float16* DIR1 = DIR0 + 16 * 32;

  const int mb = 8 * (lane >> 4), nc = lane & 15;

  for (int ray = blockIdx.x; ray < n_rays; ray += gridDim.x) {
    // ---- Harmonic embeddings: one point per lane (32 points per wave) ----
    {
      const int m = lane;
      const int g = ray * 128 + wave * 32 + m;
      float xs[3] = { pts[g * 3 + 0], pts[g * 3 + 1], pts[g * 3 + 2] };
      _Float16* row = XYZ0 + m * 64;   // rows 16..31 land in XYZ1 region
#pragma unroll
      for (int c = 0; c < 3; ++c) {
        float f = 1.f;
#pragma unroll
        for (int h = 0; h < 6; ++h) {
          float v = xs[c] * f;
          row[c * 6 + h]      = (_Float16)__sinf(v);
          row[18 + c * 6 + h] = (_Float16)__cosf(v);
          f *= 2.f;
        }
        row[36 + c] = (_Float16)xs[c];
      }
#pragma unroll
      for (int j = 39; j < 64; ++j) row[j] = (_Float16)0.f;

      float dv[3] = { dirs[ray * 3 + 0], dirs[ray * 3 + 1], dirs[ray * 3 + 2] };
      _Float16* drow = DIR0 + m * 32;
#pragma unroll
      for (int c = 0; c < 3; ++c) {
        float f = 1.f;
#pragma unroll
        for (int h = 0; h < 2; ++h) {
          float v = dv[c] * f;
          drow[c * 2 + h]     = (_Float16)__sinf(v);
          drow[6 + c * 2 + h] = (_Float16)__cosf(v);
          f *= 2.f;
        }
        drow[12 + c] = (_Float16)dv[c];
      }
#pragma unroll
      for (int j = 15; j < 32; ++j) drow[j] = (_Float16)0.f;
    }
    LDS_FENCE();

    v8f acc0[9], acc1[9];

    // layer 0: [32x64] x [64x128]
    init_bias<8>(sbias + B0_OFF, lane, acc0);
    init_bias<8>(sbias + B0_OFF, lane, acc1);
    dense2<2, 8>(XYZ0, XYZ1, 64, smem + W0T_OFF, 64, lane, acc0, acc1);
    store_relu<8>(ACT0, 128, lane, acc0);
    store_relu<8>(ACT1, 128, lane, acc1);
    LDS_FENCE();

    // layer 1
    init_bias<8>(sbias + B1_OFF, lane, acc0);
    init_bias<8>(sbias + B1_OFF, lane, acc1);
    dense2<4, 8>(ACT0, ACT1, 128, smem + W1T_OFF, 128, lane, acc0, acc1);
    store_relu<8>(ACT0, 128, lane, acc0);
    store_relu<8>(ACT1, 128, lane, acc1);
    LDS_FENCE();

    // layer 2
    init_bias<8>(sbias + B2_OFF, lane, acc0);
    init_bias<8>(sbias + B2_OFF, lane, acc1);
    dense2<4, 8>(ACT0, ACT1, 128, smem + W2T_OFF, 128, lane, acc0, acc1);
    store_relu<8>(ACT0, 128, lane, acc0);
    store_relu<8>(ACT1, 128, lane, acc1);
    LDS_FENCE();

    // layer 3 (skip concat = two accumulating matmuls)
    init_bias<8>(sbias + B3_OFF, lane, acc0);
    init_bias<8>(sbias + B3_OFF, lane, acc1);
    dense2<4, 8>(ACT0, ACT1, 128, smem + W3AT_OFF, 128, lane, acc0, acc1);
    dense2<2, 8>(XYZ0, XYZ1, 64, smem + W3BT_OFF, 64, lane, acc0, acc1);
    store_relu<8>(ACT0, 128, lane, acc0);
    store_relu<8>(ACT1, 128, lane, acc1);
    LDS_FENCE();

    // layer 4
    init_bias<8>(sbias + B4_OFF, lane, acc0);
    init_bias<8>(sbias + B4_OFF, lane, acc1);
    dense2<4, 8>(ACT0, ACT1, 128, smem + W4T_OFF, 128, lane, acc0, acc1);
    store_relu<8>(ACT0, 128, lane, acc0);
    store_relu<8>(ACT1, 128, lane, acc1);
    LDS_FENCE();

    // layer 5
    init_bias<8>(sbias + B5_OFF, lane, acc0);
    init_bias<8>(sbias + B5_OFF, lane, acc1);
    dense2<4, 8>(ACT0, ACT1, 128, smem + W5T_OFF, 128, lane, acc0, acc1);
    store_relu<8>(ACT0, 128, lane, acc0);
    store_relu<8>(ACT1, 128, lane, acc1);
    LDS_FENCE();

    // density head: [32x128] x [128x144] (129 valid cols)
    init_bias<9>(sbias + BD_OFF, lane, acc0);
    init_bias<9>(sbias + BD_OFF, lane, acc1);
    dense2<4, 9>(ACT0, ACT1, 128, smem + WDT_OFF, 128, lane, acc0, acc1);

#pragma unroll
    for (int t = 0; t < 2; ++t) {
      const v8f* acc = t ? acc1 : acc0;
      _Float16* ACTt = t ? ACT1 : ACT0;
      const int rowbase = ray * 128 + wave * 32 + t * 16;
      // density = relu(x[:,0])
      if (nc == 0) {
#pragma unroll
        for (int r = 0; r < 8; ++r) {
          float x = acc[0][r];
          out[rowbase + mb + r] = x > 0.f ? x : 0.f;
        }
      }
      // h = x[:,1:129] (no relu) -> ACT cols 0..127
#pragma unroll
      for (int n = 0; n < 9; ++n) {
        int col = n * 16 + nc;
#pragma unroll
        for (int r = 0; r < 8; ++r) {
          if (col >= 1 && col <= 128)
            ACTt[(r + mb) * 128 + (col - 1)] = (_Float16)acc[n][r];
        }
      }
    }
    LDS_FENCE();

    // color layer 1: relu(h@wr1a + dir@wr1b + br1) -> ACT cols 0..63
    init_bias<4>(sbias + BR1_OFF, lane, acc0);
    init_bias<4>(sbias + BR1_OFF, lane, acc1);
    dense2<4, 4>(ACT0, ACT1, 128, smem + WR1AT_OFF, 128, lane, acc0, acc1);
    dense2<1, 4>(DIR0, DIR1, 32, smem + WR1BT_OFF, 32, lane, acc0, acc1);
    store_relu<4>(ACT0, 128, lane, acc0);
    store_relu<4>(ACT1, 128, lane, acc1);
    LDS_FENCE();

    // color layer 2: sigmoid(h2@wr2 + br2), cols 0..2 valid
    init_bias<1>(sbias + BR2_OFF, lane, acc0);
    init_bias<1>(sbias + BR2_OFF, lane, acc1);
    dense2<2, 1>(ACT0, ACT1, 128, smem + WR2T_OFF, 64, lane, acc0, acc1);
#pragma unroll
    for (int t = 0; t < 2; ++t) {
      const v8f* acc = t ? acc1 : acc0;
      const int rowbase = ray * 128 + wave * 32 + t * 16;
      if (nc < 3) {
#pragma unroll
        for (int r = 0; r < 8; ++r) {
          float x = acc[0][r];
          float s = 1.f / (1.f + __expf(-x));
          out[ntotal + (rowbase + mb + r) * 3 + nc] = s;
        }
      }
    }
    LDS_FENCE();
  }
}

extern "C" void kernel_launch(void* const* d_in, const int* in_sizes, int n_in,
                              void* d_out, int out_size, void* d_ws, size_t ws_size,
                              hipStream_t stream) {
  const float* pts  = (const float*)d_in[0];
  const float* dirs = (const float*)d_in[1];
  const int n_rays = in_sizes[1] / 3;       // 4096
  const int ntotal = in_sizes[0] / 3;       // 524288 points

  const int from_ws = (ws_size >= (size_t)WS_BYTES) ? 1 : 0;
  if (from_ws) {
    hipLaunchKernelGGL(nerf_prep_kernel, dim3(128), dim3(256), 0, stream,
                       (const float*)d_in[2],  (const float*)d_in[3],
                       (const float*)d_in[4],  (const float*)d_in[5],
                       (const float*)d_in[6],  (const float*)d_in[7],
                       (const float*)d_in[8],  (const float*)d_in[9],
                       (const float*)d_in[10], (const float*)d_in[11],
                       (const float*)d_in[12], (const float*)d_in[13],
                       (const float*)d_in[14], (const float*)d_in[15],
                       (const float*)d_in[16], (const float*)d_in[17],
                       (const float*)d_in[18], (const float*)d_in[19],
                       d_ws);
  }

  int blocks = n_rays < 1024 ? n_rays : 1024;   // persistent: ~4 rays per block
  hipLaunchKernelGGL(nerf_fused_kernel, dim3(blocks), dim3(THREADS), 0, stream,
                     pts, dirs,
                     (const float*)d_in[2],  (const float*)d_in[3],
                     (const float*)d_in[4],  (const float*)d_in[5],
                     (const float*)d_in[6],  (const float*)d_in[7],
                     (const float*)d_in[8],  (const float*)d_in[9],
                     (const float*)d_in[10], (const float*)d_in[11],
                     (const float*)d_in[12], (const float*)d_in[13],
                     (const float*)d_in[14], (const float*)d_in[15],
                     (const float*)d_in[16], (const float*)d_in[17],
                     (const float*)d_in[18], (const float*)d_in[19],
                     (const void*)d_ws, from_ws,
                     (float*)d_out, ntotal, n_rays);
}
